// New_AdapterLayer_18442589569332
// MI455X (gfx1250) — matrix operand order repaired and checked by reading, not compiled, not run on hardware
//
#include <hip/hip_runtime.h>
#include <hip/hip_bf16.h>
#include <math.h>

// ---------------------------------------------------------------------------
// Problem constants (from reference)
// ---------------------------------------------------------------------------
#define BB    16
#define CDIM  64
#define RANK  32
#define HH    128
#define WW    128
#define HWN   (HH*WW)          // 16384
#define EE    4
#define KTOP  2
#define BINS  8
#define FREQD 64
#define KS7   7
#define NOISE_STD 0.25f

typedef __bf16 v16bf __attribute__((ext_vector_type(16)));
typedef float  v8f   __attribute__((ext_vector_type(8)));

#define FFT_PITCH 130   // 128 + 2 pad to dodge LDS bank conflicts on column pass

__device__ __forceinline__ int brev7(int v) {
    return (int)(__brev((unsigned)v) >> 25);
}

// ---------------------------------------------------------------------------
// Cooperative 128-point FFT over a 128x128 tile in LDS.
// Forward: DIF (natural in -> bit-reversed out).  Inverse: DIT (bitrev in ->
// natural out, caller applies 1/N scaling).  Whole block participates.
// ---------------------------------------------------------------------------
__device__ void fft_rows(float* re, float* im, bool inverse) {
    const float TWO_PI = 6.283185307179586f;
    if (!inverse) {
        for (int len = 128; len >= 2; len >>= 1) {
            int half = len >> 1;
            for (int q = threadIdx.x; q < 128 * 64; q += blockDim.x) {
                int row = q >> 6, j = q & 63;
                int grp = j / half, pos = j - grp * half;
                int i0 = row * FFT_PITCH + grp * len + pos;
                int i1 = i0 + half;
                float ar = re[i0], ai = im[i0], br = re[i1], bi = im[i1];
                float ang = -TWO_PI * (float)pos / (float)len;
                float s, c; __sincosf(ang, &s, &c);
                float dr = ar - br, di = ai - bi;
                re[i0] = ar + br; im[i0] = ai + bi;
                re[i1] = dr * c - di * s;
                im[i1] = dr * s + di * c;
            }
            __syncthreads();
        }
    } else {
        for (int len = 2; len <= 128; len <<= 1) {
            int half = len >> 1;
            for (int q = threadIdx.x; q < 128 * 64; q += blockDim.x) {
                int row = q >> 6, j = q & 63;
                int grp = j / half, pos = j - grp * half;
                int i0 = row * FFT_PITCH + grp * len + pos;
                int i1 = i0 + half;
                float ang = TWO_PI * (float)pos / (float)len;
                float s, c; __sincosf(ang, &s, &c);
                float br = re[i1], bi = im[i1];
                float tr = br * c - bi * s;
                float ti = br * s + bi * c;
                float ar = re[i0], ai = im[i0];
                re[i0] = ar + tr; im[i0] = ai + ti;
                re[i1] = ar - tr; im[i1] = ai - ti;
            }
            __syncthreads();
        }
    }
}

__device__ void fft_cols(float* re, float* im, bool inverse) {
    const float TWO_PI = 6.283185307179586f;
    if (!inverse) {
        for (int len = 128; len >= 2; len >>= 1) {
            int half = len >> 1;
            for (int q = threadIdx.x; q < 128 * 64; q += blockDim.x) {
                int col = q & 127, j = q >> 7;
                int grp = j / half, pos = j - grp * half;
                int i0 = (grp * len + pos) * FFT_PITCH + col;
                int i1 = i0 + half * FFT_PITCH;
                float ar = re[i0], ai = im[i0], br = re[i1], bi = im[i1];
                float ang = -TWO_PI * (float)pos / (float)len;
                float s, c; __sincosf(ang, &s, &c);
                float dr = ar - br, di = ai - bi;
                re[i0] = ar + br; im[i0] = ai + bi;
                re[i1] = dr * c - di * s;
                im[i1] = dr * s + di * c;
            }
            __syncthreads();
        }
    } else {
        for (int len = 2; len <= 128; len <<= 1) {
            int half = len >> 1;
            for (int q = threadIdx.x; q < 128 * 64; q += blockDim.x) {
                int col = q & 127, j = q >> 7;
                int grp = j / half, pos = j - grp * half;
                int i0 = (grp * len + pos) * FFT_PITCH + col;
                int i1 = i0 + half * FFT_PITCH;
                float ang = TWO_PI * (float)pos / (float)len;
                float s, c; __sincosf(ang, &s, &c);
                float br = re[i1], bi = im[i1];
                float tr = br * c - bi * s;
                float ti = br * s + bi * c;
                float ar = re[i0], ai = im[i0];
                re[i0] = ar + tr; im[i0] = ai + ti;
                re[i1] = ar - tr; im[i1] = ai - ti;
            }
            __syncthreads();
        }
    }
}

// ---------------------------------------------------------------------------
// K0: zero init (out + gate/femb accumulators)
// ---------------------------------------------------------------------------
__global__ void init_kernel(float* out, float* gates, float* femb, int n) {
    int i = blockIdx.x * blockDim.x + threadIdx.x;
    if (i < n)   out[i]  = 0.f;
    if (i < BB * EE)   gates[i] = 0.f;
    if (i < BB * BINS) femb[i]  = 0.f;
}

// ---------------------------------------------------------------------------
// K1: GAP means per (b,c)  -> means[B*C]
// ---------------------------------------------------------------------------
__global__ __launch_bounds__(256) void mean_kernel(const float* __restrict__ x,
                                                   float* __restrict__ means) {
    __shared__ float red[256];
    int bc = blockIdx.x;
    const float* src = x + (size_t)bc * HWN;
    float s = 0.f;
    for (int i = threadIdx.x; i < HWN; i += 256) s += src[i];
    red[threadIdx.x] = s;
    __syncthreads();
    for (int off = 128; off > 0; off >>= 1) {
        if (threadIdx.x < off) red[threadIdx.x] += red[threadIdx.x + off];
        __syncthreads();
    }
    if (threadIdx.x == 0) means[bc] = red[0] * (1.f / (float)HWN);
}

// ---------------------------------------------------------------------------
// K2: freq_energy: full 2D FFT of each (b,c) channel, |X|/(HW) binned into 8
//     radial bins over the (index-64) grid, mean over channels.
// ---------------------------------------------------------------------------
__global__ __launch_bounds__(256) void freq_energy_kernel(const float* __restrict__ x,
                                                          float* __restrict__ femb) {
    extern __shared__ float smem[];
    float* re   = smem;
    float* im   = smem + FFT_PITCH * 128;
    float* bins = im   + FFT_PITCH * 128;
    int bc = blockIdx.x;
    int b  = bc >> 6;              // / CDIM
    const float* src = x + (size_t)bc * HWN;
    for (int i = threadIdx.x; i < HWN; i += blockDim.x) {
        int y = i >> 7, xx = i & 127;
        re[y * FFT_PITCH + xx] = src[i];
        im[y * FFT_PITCH + xx] = 0.f;
    }
    if (threadIdx.x < BINS) bins[threadIdx.x] = 0.f;
    __syncthreads();
    fft_rows(re, im, false);
    fft_cols(re, im, false);
    const float dmax = 90.50966799187808f;      // sqrt(2)*64
    for (int i = threadIdx.x; i < HWN; i += blockDim.x) {
        int ys = i >> 7, xs = i & 127;
        int ky = brev7(ys), kx = brev7(xs);     // bit-reversed storage -> natural freq
        float vr = re[ys * FFT_PITCH + xs], vi = im[ys * FFT_PITCH + xs];
        float mag = sqrtf(vr * vr + vi * vi) * (1.f / (float)HWN);  // norm='forward'
        float dy = (float)(ky - 64), dx = (float)(kx - 64);
        float dist = sqrtf(dy * dy + dx * dx);
        int bin = (int)floorf(dist * ((float)BINS / dmax));
        if (bin < BINS) atomicAdd(&bins[bin], mag);
    }
    __syncthreads();
    if (threadIdx.x < BINS)
        atomicAdd(&femb[b * BINS + threadIdx.x], bins[threadIdx.x] * (1.f / (float)CDIM));
}

// ---------------------------------------------------------------------------
// K3: routing: img gate + freq-MLP gate -> softmax(+noise) -> top-2 gates
// ---------------------------------------------------------------------------
__global__ void routing_kernel(const float* __restrict__ means,
                               const float* __restrict__ femb,
                               const float* __restrict__ gate_w,
                               const float* __restrict__ fg_w1,
                               const float* __restrict__ fg_b1,
                               const float* __restrict__ fg_w2,
                               const float* __restrict__ noise,
                               float* __restrict__ gates) {
    int b = threadIdx.x;
    if (b >= BB) return;
    float logit[EE];
    for (int e = 0; e < EE; ++e) {
        float l = 0.f;
        for (int c = 0; c < CDIM; ++c) l += means[b * CDIM + c] * gate_w[e * CDIM + c];
        logit[e] = l;
    }
    for (int f = 0; f < FREQD; ++f) {
        float h = fg_b1[f];
        for (int k = 0; k < BINS; ++k) h += femb[b * BINS + k] * fg_w1[f * BINS + k];
        h = fmaxf(h, 0.f);
        for (int e = 0; e < EE; ++e) logit[e] += h * fg_w2[e * FREQD + f];
    }
    for (int e = 0; e < EE; ++e) logit[e] += noise[b * EE + e] * NOISE_STD;
    float m = logit[0];
    for (int e = 1; e < EE; ++e) m = fmaxf(m, logit[e]);
    float sc[EE], sum = 0.f;
    for (int e = 0; e < EE; ++e) { sc[e] = __expf(logit[e] - m); sum += sc[e]; }
    for (int e = 0; e < EE; ++e) sc[e] /= sum;
    int i1 = 0;
    for (int e = 1; e < EE; ++e) if (sc[e] > sc[i1]) i1 = e;
    int i2 = -1;
    for (int e = 0; e < EE; ++e) {
        if (e == i1) continue;
        if (i2 < 0 || sc[e] > sc[i2]) i2 = e;
    }
    for (int e = 0; e < EE; ++e)
        gates[b * EE + e] = (e == i1 || e == i2) ? sc[e] : 0.f;
}

// ---------------------------------------------------------------------------
// K4: WMMA projection: h = W0 x, g = silu(W1 x) per active (b,e).
//     GEMM  D[RANK, pix] = W[RANK, CDIM] * X[CDIM, pix]
//     bf16 A/B, f32 accumulate, v_wmma_f32_16x16x32_bf16.
//     Block = 128 threads (4 waves), each wave owns 16 pixels; grid (HW/64, B).
// ---------------------------------------------------------------------------
__global__ __launch_bounds__(128) void proj_kernel(const float* __restrict__ x,
                                                   const float* __restrict__ proj0,
                                                   const float* __restrict__ proj1,
                                                   const float* __restrict__ gates,
                                                   int e,
                                                   float* __restrict__ hbuf,
                                                   float* __restrict__ gbuf) {
    int b = blockIdx.y;
    if (gates[b * EE + e] == 0.f) return;          // uniform per block
    int wave  = threadIdx.x >> 5;
    int lane  = threadIdx.x & 31;
    int nlane = lane & 15;
    int hi    = lane >> 4;
    int pix0  = blockIdx.x * 64 + wave * 16;
    int p     = pix0 + nlane;
    const float* xb = x + (size_t)b * CDIM * HWN;

    v8f acc0[2] = {v8f{}, v8f{}};   // h accumulators, M-tiles 0..1
    v8f acc1[2] = {v8f{}, v8f{}};   // g accumulators

    #pragma unroll
    for (int kk = 0; kk < 2; ++kk) {               // K = 64 -> two 32-wide steps
        // B fragment (32x16): lanes 0-15 hold K=0..15, lanes 16-31 K=16..31
        v16bf bfrag;
        #pragma unroll
        for (int i = 0; i < 16; ++i) {
            int c = kk * 32 + hi * 16 + i;
            bfrag[i] = (__bf16)xb[(size_t)c * HWN + p];
        }
        // prefetch next batch-row of x (lowers to global_prefetch_b8)
        __builtin_prefetch(&xb[(size_t)(kk * 32 + hi * 16) * HWN + p + 256], 0, 1);
        #pragma unroll
        for (int mt = 0; mt < 2; ++mt) {           // RANK = 32 -> two M tiles
            int r = mt * 16 + nlane;
            v16bf a0, a1;
            #pragma unroll
            for (int i = 0; i < 16; ++i) {
                int kl = hi * 16 + i;
                int c  = kk * 32 + kl;
                a0[i] = (__bf16)proj0[(size_t)(e * RANK + r) * CDIM + c];
                a1[i] = (__bf16)proj1[(size_t)(e * RANK + r) * CDIM + c];
            }
            acc0[mt] = __builtin_amdgcn_wmma_f32_16x16x32_bf16(
                false, a0, false, bfrag, (short)0, acc0[mt], false, false);
            acc1[mt] = __builtin_amdgcn_wmma_f32_16x16x32_bf16(
                false, a1, false, bfrag, (short)0, acc1[mt], false, false);
        }
    }
    float* hch = hbuf + (size_t)b * RANK * HWN;
    float* gch = gbuf + (size_t)b * RANK * HWN;
    #pragma unroll
    for (int mt = 0; mt < 2; ++mt) {
        #pragma unroll
        for (int v = 0; v < 8; ++v) {
            int r = mt * 16 + v + hi * 8;          // f32 C/D layout
            float hv = acc0[mt][v];
            float gv = acc1[mt][v];
            hch[(size_t)r * HWN + p] = hv;
            gch[(size_t)r * HWN + p] = gv / (1.f + __expf(-gv));   // silu
        }
    }
}

// ---------------------------------------------------------------------------
// K5: FFT radial filter body (experts 0 and 2), in-place on h channel.
//     mag*filt*exp(i*phase) == filt * X  -> real mask on the spectrum.
// ---------------------------------------------------------------------------
__global__ __launch_bounds__(256) void fft_filter_kernel(float* __restrict__ hbuf,
                                                         const float* __restrict__ gates,
                                                         int e,
                                                         const float* __restrict__ gain_p,
                                                         const float* __restrict__ decay_p,
                                                         float cmax) {
    int br = blockIdx.x;
    int b  = br >> 5;               // / RANK
    if (gates[b * EE + e] == 0.f) return;
    extern __shared__ float smem[];
    float* re = smem;
    float* im = smem + FFT_PITCH * 128;
    float gain  = gain_p[0];
    float decay = decay_p[0];
    float* ch = hbuf + (size_t)br * HWN;
    for (int i = threadIdx.x; i < HWN; i += blockDim.x) {
        int y = i >> 7, xx = i & 127;
        re[y * FFT_PITCH + xx] = ch[i];
        im[y * FFT_PITCH + xx] = 0.f;
    }
    __syncthreads();
    fft_rows(re, im, false);
    fft_cols(re, im, false);
    // multiply by radial filter (bit-reversed storage domain)
    for (int i = threadIdx.x; i < HWN; i += blockDim.x) {
        int ys = i >> 7, xs = i & 127;
        int ky = brev7(ys), kx = brev7(xs);
        float fy = (float)((ky < 64) ? ky : ky - 128) * (1.f / 128.f);
        float fx = (float)((kx < 64) ? kx : kx - 128) * (1.f / 128.f);
        float fg = sqrtf(fy * fy + fx * fx);
        float filt = (1.f - __expf(-gain * fg)) * __expf(-decay * fg);
        filt = fminf(fmaxf(filt, 0.f), cmax);
        int idx = ys * FFT_PITCH + xs;
        re[idx] *= filt;
        im[idx] *= filt;
    }
    __syncthreads();
    fft_cols(re, im, true);
    fft_rows(re, im, true);
    for (int i = threadIdx.x; i < HWN; i += blockDim.x) {
        int y = i >> 7, xx = i & 127;
        ch[i] = re[y * FFT_PITCH + xx] * (1.f / (float)HWN);  // 1/128 per dim
    }
}

// ---------------------------------------------------------------------------
// K6: depthwise conv (3x3 or 7x7) with optional exact GELU epilogue
// ---------------------------------------------------------------------------
__global__ __launch_bounds__(256) void dwconv_kernel(const float* __restrict__ in,
                                                     float* __restrict__ out,
                                                     const float* __restrict__ w,
                                                     const float* __restrict__ bias,
                                                     const float* __restrict__ gates,
                                                     int e, int ksize, int dogelu) {
    int idx = blockIdx.x * blockDim.x + threadIdx.x;        // over B*RANK*HW
    int b   = idx >> 19;                                     // / (RANK*HW)
    if (gates[b * EE + e] == 0.f) return;                    // uniform per block
    int r   = (idx >> 14) & 31;
    int pix = idx & (HWN - 1);
    int y = pix >> 7, xx = pix & 127;
    int pad = ksize >> 1;
    const float* ch = in + (size_t)(b * RANK + r) * HWN;
    float sum = bias[r];
    for (int ky = 0; ky < ksize; ++ky) {
        int iy = y + ky - pad;
        if (iy < 0 || iy >= HH) continue;
        for (int kx = 0; kx < ksize; ++kx) {
            int ix = xx + kx - pad;
            if (ix < 0 || ix >= WW) continue;
            sum += ch[iy * WW + ix] * w[(r * ksize + ky) * ksize + kx];
        }
    }
    if (dogelu) sum = 0.5f * sum * (1.f + erff(sum * 0.70710678118654752f));
    out[idx] = sum;
}

// ---------------------------------------------------------------------------
// K7: AvgPool2d(3, stride 1, pad 1, count_include_pad) == 3x3 sum / 9
// ---------------------------------------------------------------------------
__global__ __launch_bounds__(256) void avgpool_kernel(const float* __restrict__ in,
                                                      float* __restrict__ out,
                                                      const float* __restrict__ gates,
                                                      int e) {
    int idx = blockIdx.x * blockDim.x + threadIdx.x;
    int b   = idx >> 19;
    if (gates[b * EE + e] == 0.f) return;
    int r   = (idx >> 14) & 31;
    int pix = idx & (HWN - 1);
    int y = pix >> 7, xx = pix & 127;
    const float* ch = in + (size_t)(b * RANK + r) * HWN;
    float sum = 0.f;
    for (int ky = -1; ky <= 1; ++ky) {
        int iy = y + ky;
        if (iy < 0 || iy >= HH) continue;
        for (int kx = -1; kx <= 1; ++kx) {
            int ix = xx + kx;
            if (ix < 0 || ix >= WW) continue;
            sum += ch[iy * WW + ix];
        }
    }
    out[idx] = sum * (1.f / 9.f);
}

// ---------------------------------------------------------------------------
// K8: WMMA combine: out += gate * ( proj2 * (h_body .* g) + x )
//     GEMM  D[CDIM, pix] = W2[CDIM, RANK] * (h.*g)[RANK, pix]  (single K step)
// ---------------------------------------------------------------------------
__global__ __launch_bounds__(128) void combine_kernel(const float* __restrict__ x,
                                                      const float* __restrict__ proj2,
                                                      const float* __restrict__ gates,
                                                      int e,
                                                      const float* __restrict__ hbuf,
                                                      const float* __restrict__ gbuf,
                                                      float* __restrict__ out) {
    int b = blockIdx.y;
    float gate = gates[b * EE + e];
    if (gate == 0.f) return;
    int wave  = threadIdx.x >> 5;
    int lane  = threadIdx.x & 31;
    int nlane = lane & 15;
    int hi    = lane >> 4;
    int p     = blockIdx.x * 64 + wave * 16 + nlane;

    const float* hch = hbuf + (size_t)b * RANK * HWN;
    const float* gch = gbuf + (size_t)b * RANK * HWN;
    v16bf bfrag;                                   // K = r = 0..31
    #pragma unroll
    for (int i = 0; i < 16; ++i) {
        int r = hi * 16 + i;
        float v = hch[(size_t)r * HWN + p] * gch[(size_t)r * HWN + p];
        bfrag[i] = (__bf16)v;
    }
    const float* xb = x   + (size_t)b * CDIM * HWN;
    float*       ob = out + (size_t)b * CDIM * HWN;
    #pragma unroll
    for (int mt = 0; mt < 4; ++mt) {               // CDIM = 64 -> four M tiles
        int c = mt * 16 + nlane;
        v16bf afrag;
        #pragma unroll
        for (int i = 0; i < 16; ++i) {
            int r = hi * 16 + i;
            afrag[i] = (__bf16)proj2[(size_t)(e * CDIM + c) * RANK + r];
        }
        v8f acc = {};
        acc = __builtin_amdgcn_wmma_f32_16x16x32_bf16(
            false, afrag, false, bfrag, (short)0, acc, false, false);
        #pragma unroll
        for (int v = 0; v < 8; ++v) {
            int cc = mt * 16 + v + hi * 8;
            size_t off = (size_t)cc * HWN + p;
            ob[off] += gate * (acc[v] + xb[off]);
        }
    }
}

// ---------------------------------------------------------------------------
// Host-side orchestration
// ---------------------------------------------------------------------------
extern "C" void kernel_launch(void* const* d_in, const int* in_sizes, int n_in,
                              void* d_out, int out_size, void* d_ws, size_t ws_size,
                              hipStream_t stream) {
    const float* x      = (const float*)d_in[0];
    const float* noise  = (const float*)d_in[1];
    const float* gate_w = (const float*)d_in[2];
    const float* fg_w1  = (const float*)d_in[3];
    const float* fg_b1  = (const float*)d_in[4];
    const float* fg_w2  = (const float*)d_in[5];
    const float* proj0  = (const float*)d_in[6];
    const float* proj1  = (const float*)d_in[7];
    const float* proj2  = (const float*)d_in[8];
    const float* hg_gain  = (const float*)d_in[9];
    const float* hg_decay = (const float*)d_in[10];
    const float* hf_w1  = (const float*)d_in[11];
    const float* hf_b1  = (const float*)d_in[12];
    const float* hf_w2  = (const float*)d_in[13];
    const float* hf_b2  = (const float*)d_in[14];
    const float* lg_gain  = (const float*)d_in[15];
    const float* lg_decay = (const float*)d_in[16];
    const float* lf_w   = (const float*)d_in[17];
    const float* lf_b   = (const float*)d_in[18];
    float* out = (float*)d_out;

    // workspace layout (floats)
    float* ws = (float*)d_ws;
    float* gates = ws;                 // 64
    float* femb  = ws + 64;            // 128
    float* means = ws + 192;           // 1024
    const size_t CH = (size_t)BB * RANK * HWN;   // 8,388,608 floats per buffer
    float* hbuf = ws + 4096;
    float* gbuf = hbuf + CH;
    float* tbuf = gbuf + CH;

    const int NOUT = BB * CDIM * HWN;                 // 16,777,216
    const size_t FFT_SMEM = (size_t)(2 * FFT_PITCH * 128 + 16) * sizeof(float);

    // 0) zero accumulators + output
    init_kernel<<<(NOUT + 255) / 256, 256, 0, stream>>>(out, gates, femb, NOUT);
    // 1) GAP means
    mean_kernel<<<BB * CDIM, 256, 0, stream>>>(x, means);
    // 2) frequency-energy histogram (2D FFT per channel in LDS)
    freq_energy_kernel<<<BB * CDIM, 256, FFT_SMEM, stream>>>(x, femb);
    // 3) routing -> sparse top-2 gates
    routing_kernel<<<1, 32, 0, stream>>>(means, femb, gate_w, fg_w1, fg_b1,
                                         fg_w2, noise, gates);

    dim3 gemmGrid(HWN / 64, BB);
    const int convBlocks = (BB * RANK * HWN) / 256;   // 32768

    for (int e = 0; e < EE; ++e) {
        // projections (skips inactive batches via gate==0)
        proj_kernel<<<gemmGrid, 128, 0, stream>>>(x, proj0, proj1, gates, e,
                                                  hbuf, gbuf);
        // expert body -> result lands back in hbuf
        if (e == 0) {
            fft_filter_kernel<<<BB * RANK, 256, FFT_SMEM, stream>>>(
                hbuf, gates, e, hg_gain, hg_decay, 3.0f);
        } else if (e == 1) {
            dwconv_kernel<<<convBlocks, 256, 0, stream>>>(hbuf, tbuf, hf_w1,
                                                          hf_b1, gates, e, 3, 1);
            dwconv_kernel<<<convBlocks, 256, 0, stream>>>(tbuf, hbuf, hf_w2,
                                                          hf_b2, gates, e, 3, 0);
        } else if (e == 2) {
            fft_filter_kernel<<<BB * RANK, 256, FFT_SMEM, stream>>>(
                hbuf, gates, e, lg_gain, lg_decay, 1.0f);
        } else {
            dwconv_kernel<<<convBlocks, 256, 0, stream>>>(hbuf, tbuf, lf_w,
                                                          lf_b, gates, e, KS7, 1);
            avgpool_kernel<<<convBlocks, 256, 0, stream>>>(tbuf, hbuf, gates, e);
        }
        // combine: out += gate * (proj2 * (h .* g) + x)
        combine_kernel<<<gemmGrid, 128, 0, stream>>>(x, proj2, gates, e,
                                                     hbuf, gbuf, out);
    }
    (void)in_sizes; (void)n_in; (void)out_size; (void)ws_size;
}
